// Generator_18648747999715
// MI455X (gfx1250) — compile-verified, matrix operand order checked by reading, not run on hardware
//
#include <hip/hip_runtime.h>
#include <math.h>

// ---------------------------------------------------------------------------
// GraphRNN decoder loss on MI455X (gfx1250), single persistent workgroup.
// All matmuls via v_wmma_f32_16x16x32_f16 (f16 in, f32 accumulate).
// Weights pre-transposed to [N][K] f16 so every WMMA fragment is two b128
// loads per lane matching the CDNA5 16-bit A/B VGPR layout.
// All device-memory accesses are forced into address_space(1) so the backend
// emits global_load/global_store (LOADcnt only) instead of flat ops
// (which would also tie up DScnt and need 64-bit VGPR address pairs).
// ---------------------------------------------------------------------------

typedef _Float16 f16;
typedef f16   half8 __attribute__((ext_vector_type(8)));
typedef f16   v16h  __attribute__((ext_vector_type(16)));
typedef float v8f   __attribute__((ext_vector_type(8)));

typedef __attribute__((address_space(1))) float g_f32;
typedef __attribute__((address_space(1))) f16   g_f16;
typedef __attribute__((address_space(1))) half8 g_h8;
typedef __attribute__((address_space(1))) int   g_i32;

#define HD    256
#define H4    1024
#define BATCH 32
#define TT    16
#define SZ    (BATCH * HD)   // one layer of state: 8192 elements

union Frag16 { v16h v; half8 h[2]; };

// Load one 16x32 f16 WMMA fragment (A or B) for this lane.
// layout: lanes 0-15 row=lane, K in VGPR0-3 = half*8..+7, VGPR4-7 = +16.
static __device__ inline v16h load_frag(const f16* base, int row, int ldk,
                                        int k0, int half) {
  Frag16 f;
  const g_h8* g =
      (const g_h8*)(base + (size_t)row * ldk + k0 + half * 8);
  f.h[0] = g[0];
  f.h[1] = g[2];  // +16 f16 elements
  return f.v;
}

static __device__ inline float sigm(float x) {
  return 1.0f / (1.0f + __expf(-x));
}

// cross-wave visibility of global writes within the workgroup
static __device__ inline void gsync() {
  __builtin_amdgcn_fence(__ATOMIC_RELEASE, "agent");
  __syncthreads();
  __builtin_amdgcn_fence(__ATOMIC_ACQUIRE, "agent");
}

// ---------------------------------------------------------------------------
// One LSTM layer:  gates = Xin[32,Kin]@Wi + Hst[32,256]@Wh + bsum
// Weight matrices are pre-transposed f16: WiT[1024][Kin], WhT[1024][256].
// Hst is updated in place (internal barrier separates read/write phases).
// Cst is f32 cell state. Optionally writes (f16)c2 to cCopy (ectx feed).
// 16 waves: wave w owns columns [16w,16w+16) of H for all four gates.
// ---------------------------------------------------------------------------
static __device__ void lstm_stage(const f16* Xin, int Kin, const f16* WiT,
                                  f16* Hst, const f16* WhT,
                                  const float* bsum, float* Cst,
                                  f16* cCopy, int cLd, int cOff) {
  const int tid  = threadIdx.x;
  const int wave = tid >> 5;
  const int lane = tid & 31;
  const int nLoc = lane & 15;
  const int half = lane >> 4;
  const int n    = wave * 16 + nLoc;

  v8f acc[2][4];
#pragma unroll
  for (int mt = 0; mt < 2; ++mt)
#pragma unroll
    for (int g = 0; g < 4; ++g)
#pragma unroll
      for (int r = 0; r < 8; ++r) acc[mt][g][r] = 0.0f;

  const int kinTiles = Kin >> 5;
  for (int kt = 0; kt < kinTiles; ++kt) {
    v16h a0 = load_frag(Xin, nLoc,      Kin, kt * 32, half);
    v16h a1 = load_frag(Xin, 16 + nLoc, Kin, kt * 32, half);
#pragma unroll
    for (int g = 0; g < 4; ++g) {
      v16h b = load_frag(WiT, g * HD + n, Kin, kt * 32, half);
      acc[0][g] = __builtin_amdgcn_wmma_f32_16x16x32_f16(
          false, a0, false, b, (short)0, acc[0][g], false, false);
      acc[1][g] = __builtin_amdgcn_wmma_f32_16x16x32_f16(
          false, a1, false, b, (short)0, acc[1][g], false, false);
    }
  }
#pragma unroll
  for (int kt = 0; kt < 8; ++kt) {
    v16h a0 = load_frag(Hst, nLoc,      HD, kt * 32, half);
    v16h a1 = load_frag(Hst, 16 + nLoc, HD, kt * 32, half);
#pragma unroll
    for (int g = 0; g < 4; ++g) {
      v16h b = load_frag(WhT, g * HD + n, HD, kt * 32, half);
      acc[0][g] = __builtin_amdgcn_wmma_f32_16x16x32_f16(
          false, a0, false, b, (short)0, acc[0][g], false, false);
      acc[1][g] = __builtin_amdgcn_wmma_f32_16x16x32_f16(
          false, a1, false, b, (short)0, acc[1][g], false, false);
    }
  }

  __syncthreads();  // all waves done reading Hst before in-place overwrite

  const g_f32* bs = (const g_f32*)bsum;
  g_f32* C  = (g_f32*)Cst;
  g_f16* H  = (g_f16*)Hst;
  g_f16* CC = (g_f16*)cCopy;
  const float bI = bs[n];
  const float bF = bs[HD + n];
  const float bG = bs[2 * HD + n];
  const float bO = bs[3 * HD + n];
#pragma unroll
  for (int mt = 0; mt < 2; ++mt) {
#pragma unroll
    for (int r = 0; r < 8; ++r) {
      // D layout: VGPR r holds M=r (lanes 0-15) / M=r+8 (lanes 16-31)
      const int m  = mt * 16 + half * 8 + r;
      const float iv = acc[mt][0][r] + bI;
      const float fv = acc[mt][1][r] + bF;
      const float gv = acc[mt][2][r] + bG;
      const float ov = acc[mt][3][r] + bO;
      const float cO = C[m * HD + n];
      const float c2 = sigm(fv) * cO + sigm(iv) * tanhf(gv);
      const float h2 = sigm(ov) * tanhf(c2);
      C[m * HD + n] = c2;
      H[m * HD + n] = (f16)h2;
      if (cCopy) CC[m * cLd + cOff + n] = (f16)c2;
    }
  }
}

// emb[32,E] = Hin[32,256] @ OutT^T + bout   (E = 128 or 64; OutT is [E][256])
static __device__ void proj_stage(const f16* Hin, const f16* OutT,
                                  const float* bout, int E,
                                  float* embF32, f16* embF16) {
  const int tid  = threadIdx.x;
  const int wave = tid >> 5;
  const int lane = tid & 31;
  const int tilesN = E >> 4;
  if (wave >= 2 * tilesN) return;  // no barrier below: early-out is safe
  const int mt   = wave / tilesN;
  const int nt   = wave % tilesN;
  const int nLoc = lane & 15;
  const int half = lane >> 4;
  const int n    = nt * 16 + nLoc;

  v8f acc;
#pragma unroll
  for (int r = 0; r < 8; ++r) acc[r] = 0.0f;
#pragma unroll
  for (int kt = 0; kt < 8; ++kt) {
    v16h a = load_frag(Hin, mt * 16 + nLoc, HD, kt * 32, half);
    v16h b = load_frag(OutT, n, HD, kt * 32, half);
    acc = __builtin_amdgcn_wmma_f32_16x16x32_f16(
        false, a, false, b, (short)0, acc, false, false);
  }
  const float bb = ((const g_f32*)bout)[n];
  g_f32* e32 = (g_f32*)embF32;
  g_f16* e16 = (g_f16*)embF16;
#pragma unroll
  for (int r = 0; r < 8; ++r) {
    const int m = mt * 16 + half * 8 + r;
    const float v = acc[r] + bb;
    e32[m * E + n] = v;
    if (embF16) e16[m * E + n] = (f16)v;
  }
}

// masked-mean cross-entropy; executed by threads 0..31 (wave 0), lane = batch
static __device__ float ce32(const float* emb, int E, const float* W,
                             const float* bias, int nc,
                             const int* tgtBase, int tStride) {
  const int b = threadIdx.x;
  const g_f32* eg = (const g_f32*)emb;
  const g_f32* Wg = (const g_f32*)W;
  const g_f32* bg = (const g_f32*)bias;
  const int t = ((const g_i32*)tgtBase)[b * tStride];
  float logits[40];
  float mx = -3.0e38f;
  for (int c = 0; c < nc; ++c) {
    float s = bg[c];
    for (int e = 0; e < E; ++e) s += eg[b * E + e] * Wg[e * nc + c];
    logits[c] = s;
    mx = fmaxf(mx, s);
  }
  float se = 0.0f;
  for (int c = 0; c < nc; ++c) se += __expf(logits[c] - mx);
  const int valid = (t >= 0) ? 1 : 0;
  const int tt = valid ? t : 0;
  const float nll = (__logf(se) + mx) - logits[tt];
  float s = valid ? nll : 0.0f;
  int cnt = valid;
  for (int off = 16; off > 0; off >>= 1) {
    s   += __shfl_xor(s, off, 32);
    cnt += __shfl_xor(cnt, off, 32);
  }
  return s / (float)(cnt > 1 ? cnt : 1);
}

struct Params {
  const f16 *nWi0T, *nWh0T, *nOutT;
  const f16 *nWiT[3], *nWhT[3];
  const f16 *eWi0T, *eWh0T, *eOutT;
  const f16 *eWiT[3], *eWhT[3];
  const float *nb[4], *eb[4];
  const float *nOutb, *eOutb;
  const float *ncW0, *ncb0, *ncW1, *ncb1, *ncW2, *ncb2;
  const float *ecW0, *ecb0, *ecW1, *ecb1;
  const int *atomT, *bondT;
  f16 *nodeX0;   // [32][512]  (cols 256..511 = ctx, prefilled by prep)
  f16 *edgeX0;   // [32][384]  (cols 128..383 = ectx, written by node layer 3)
  f16 *nodeH;    // [4][32][256]
  f16 *edgeH;    // [4][32][256]
  f16 *bufE;     // [16][32][128] node-embedding ring
  float *nodeC;  // [4][32][256]
  float *edgeC;  // [4][32][256]
  float *nodeEmb;  // [32][128]
  float *edgeEmb;  // [32][64]
  float *outLoss;
};

__global__ __launch_bounds__(512, 1) void grnn_main(Params P) {
  const int tid = threadIdx.x;

  g_f16* nodeHg = (g_f16*)P.nodeH;
  g_f32* nodeCg = (g_f32*)P.nodeC;
  g_f16* edgeHg = (g_f16*)P.edgeH;
  g_f32* edgeCg = (g_f32*)P.edgeC;
  g_f16* nodeX0g = (g_f16*)P.nodeX0;
  g_f16* edgeX0g = (g_f16*)P.edgeX0;
  const g_f16* bufEg = (const g_f16*)P.bufE;

  // zero node state + x-part of nodeX0 (ctx half stays)
  for (int idx = tid; idx < 4 * SZ; idx += 512) {
    nodeHg[idx] = (f16)0.0f;
    nodeCg[idx] = 0.0f;
  }
  for (int idx = tid; idx < SZ; idx += 512) {
    const int m = idx >> 8, n = idx & 255;
    nodeX0g[m * 512 + n] = (f16)0.0f;
  }
  float loss = 0.0f;
  gsync();

  for (int i = 0; i < TT; ++i) {
    // ---- node cell ----
    lstm_stage(P.nodeX0, 512, P.nWi0T, P.nodeH, P.nWh0T, P.nb[0], P.nodeC,
               nullptr, 0, 0);
    gsync();
    lstm_stage(P.nodeH, 256, P.nWiT[0], P.nodeH + SZ, P.nWhT[0], P.nb[1],
               P.nodeC + SZ, nullptr, 0, 0);
    gsync();
    lstm_stage(P.nodeH + SZ, 256, P.nWiT[1], P.nodeH + 2 * SZ, P.nWhT[1],
               P.nb[2], P.nodeC + 2 * SZ, nullptr, 0, 0);
    gsync();
    // layer 3 also emits ectx = (f16)c2 into edgeX0 cols 128..383
    lstm_stage(P.nodeH + 2 * SZ, 256, P.nWiT[2], P.nodeH + 3 * SZ, P.nWhT[2],
               P.nb[3], P.nodeC + 3 * SZ, P.edgeX0, 384, 128);
    gsync();
    proj_stage(P.nodeH + 3 * SZ, P.nOutT, P.nOutb, 128, P.nodeEmb,
               P.bufE + (size_t)i * BATCH * 128);
    gsync();
    if (tid < 32) {
      loss += ce32(P.nodeEmb, 128, P.ncW0, P.ncb0, 40,
                   P.atomT + i * BATCH * 3 + 0, 3);
      loss += ce32(P.nodeEmb, 128, P.ncW1, P.ncb1, 6,
                   P.atomT + i * BATCH * 3 + 1, 3);
      loss += ce32(P.nodeEmb, 128, P.ncW2, P.ncb2, 5,
                   P.atomT + i * BATCH * 3 + 2, 3);
    }

    // ---- edge scan: j >= i is fully masked in the reference -> skip ----
    for (int idx = tid; idx < 4 * SZ; idx += 512) {
      edgeHg[idx] = (f16)0.0f;
      edgeCg[idx] = 0.0f;
    }
    gsync();
    for (int j = i - 1; j >= 0; --j) {
      for (int idx = tid; idx < BATCH * 128; idx += 512) {
        const int m = idx >> 7, n = idx & 127;
        edgeX0g[m * 384 + n] = bufEg[((size_t)j * BATCH + m) * 128 + n];
      }
      gsync();
      lstm_stage(P.edgeX0, 384, P.eWi0T, P.edgeH, P.eWh0T, P.eb[0], P.edgeC,
                 nullptr, 0, 0);
      gsync();
      lstm_stage(P.edgeH, 256, P.eWiT[0], P.edgeH + SZ, P.eWhT[0], P.eb[1],
                 P.edgeC + SZ, nullptr, 0, 0);
      gsync();
      lstm_stage(P.edgeH + SZ, 256, P.eWiT[1], P.edgeH + 2 * SZ, P.eWhT[1],
                 P.eb[2], P.edgeC + 2 * SZ, nullptr, 0, 0);
      gsync();
      lstm_stage(P.edgeH + 2 * SZ, 256, P.eWiT[2], P.edgeH + 3 * SZ,
                 P.eWhT[2], P.eb[3], P.edgeC + 3 * SZ, nullptr, 0, 0);
      gsync();
      proj_stage(P.edgeH + 3 * SZ, P.eOutT, P.eOutb, 64, P.edgeEmb, nullptr);
      gsync();
      if (tid < 32) {
        loss += ce32(P.edgeEmb, 64, P.ecW0, P.ecb0, 5,
                     P.bondT + i * 1024 + j * 2 + 0, 32);
        loss += ce32(P.edgeEmb, 64, P.ecW1, P.ecb1, 4,
                     P.bondT + i * 1024 + j * 2 + 1, 32);
      }
    }
    // x_{i+1} = edge c state of last layer
    for (int idx = tid; idx < SZ; idx += 512) {
      const int m = idx >> 8, n = idx & 255;
      nodeX0g[m * 512 + n] = (f16)edgeCg[3 * SZ + m * HD + n];
    }
    gsync();
  }
  if (tid == 0) ((g_f32*)P.outLoss)[0] = loss;
}

// ----------------------------- prep kernels --------------------------------
__global__ void k_transpose(const float* src, f16* dst, int K, int N) {
  const int idx = blockIdx.x * 256 + threadIdx.x;
  if (idx >= K * N) return;
  const int k = idx / N, n = idx % N;
  ((g_f16*)dst)[(size_t)n * K + k] = (f16)((const g_f32*)src)[idx];
}

__global__ void k_bias_sum(const float* a, const float* b, float* o, int n) {
  const int i = blockIdx.x * 256 + threadIdx.x;
  if (i < n)
    ((g_f32*)o)[i] = ((const g_f32*)a)[i] + ((const g_f32*)b)[i];
}

// ctx = z @ latent_W + latent_b, written as f16 into nodeX0 cols 256..511
__global__ void k_ctx(const float* z, const float* W, const float* b,
                      f16* nodeX0) {
  const int idx = blockIdx.x * 256 + threadIdx.x;
  if (idx >= 32 * 256) return;
  const int bb = idx >> 8, n = idx & 255;
  const g_f32* zg = (const g_f32*)z;
  const g_f32* Wg = (const g_f32*)W;
  float s = ((const g_f32*)b)[n];
  for (int k = 0; k < 128; ++k) s += zg[bb * 128 + k] * Wg[k * 256 + n];
  ((g_f16*)nodeX0)[bb * 512 + 256 + n] = (f16)s;
}

// ----------------------------- host side -----------------------------------
extern "C" void kernel_launch(void* const* d_in, const int* in_sizes, int n_in,
                              void* d_out, int out_size, void* d_ws,
                              size_t ws_size, hipStream_t stream) {
  (void)in_sizes; (void)n_in; (void)out_size; (void)ws_size;
  auto F = [&](int i) { return (const float*)d_in[i]; };
  auto I = [&](int i) { return (const int*)d_in[i]; };

  size_t off = 0;
  char* ws = (char*)d_ws;
  auto alloc = [&](size_t bytes) -> void* {
    void* p = ws + off;
    off += (bytes + 255) & ~(size_t)255;
    return p;
  };

  Params P;
  f16* nWi0T = (f16*)alloc(1024 * 512 * 2);
  f16* nWh0T = (f16*)alloc(1024 * 256 * 2);
  f16* nWiT[3]; f16* nWhT[3]; f16* eWiT[3]; f16* eWhT[3];
  for (int l = 0; l < 3; ++l) nWiT[l] = (f16*)alloc(1024 * 256 * 2);
  for (int l = 0; l < 3; ++l) nWhT[l] = (f16*)alloc(1024 * 256 * 2);
  f16* nOutT = (f16*)alloc(128 * 256 * 2);
  f16* eWi0T = (f16*)alloc(1024 * 384 * 2);
  f16* eWh0T = (f16*)alloc(1024 * 256 * 2);
  for (int l = 0; l < 3; ++l) eWiT[l] = (f16*)alloc(1024 * 256 * 2);
  for (int l = 0; l < 3; ++l) eWhT[l] = (f16*)alloc(1024 * 256 * 2);
  f16* eOutT = (f16*)alloc(64 * 256 * 2);
  float* nb[4]; float* eb[4];
  for (int l = 0; l < 4; ++l) nb[l] = (float*)alloc(1024 * 4);
  for (int l = 0; l < 4; ++l) eb[l] = (float*)alloc(1024 * 4);
  f16* nodeX0 = (f16*)alloc(32 * 512 * 2);
  f16* edgeX0 = (f16*)alloc(32 * 384 * 2);
  f16* nodeH  = (f16*)alloc(4 * SZ * 2);
  f16* edgeH  = (f16*)alloc(4 * SZ * 2);
  f16* bufE   = (f16*)alloc(16 * 32 * 128 * 2);
  float* nodeC   = (float*)alloc(4 * SZ * 4);
  float* edgeC   = (float*)alloc(4 * SZ * 4);
  float* nodeEmb = (float*)alloc(32 * 128 * 4);
  float* edgeEmb = (float*)alloc(32 * 64 * 4);

  auto tr = [&](const float* src, f16* dst, int K, int N) {
    const int tot = K * N;
    k_transpose<<<(tot + 255) / 256, 256, 0, stream>>>(src, dst, K, N);
  };
  auto bs = [&](const float* a, const float* b, float* o) {
    k_bias_sum<<<4, 256, 0, stream>>>(a, b, o, 1024);
  };

  // weight transposes (f32 [K][N] -> f16 [N][K])
  tr(F(3), nWi0T, 512, 1024);
  tr(F(4), nWh0T, 256, 1024);
  for (int l = 0; l < 3; ++l) tr(F(7) + (size_t)l * 256 * 1024, nWiT[l], 256, 1024);
  for (int l = 0; l < 3; ++l) tr(F(8) + (size_t)l * 256 * 1024, nWhT[l], 256, 1024);
  tr(F(11), nOutT, 256, 128);
  tr(F(13), eWi0T, 384, 1024);
  tr(F(14), eWh0T, 256, 1024);
  for (int l = 0; l < 3; ++l) tr(F(17) + (size_t)l * 256 * 1024, eWiT[l], 256, 1024);
  for (int l = 0; l < 3; ++l) tr(F(18) + (size_t)l * 256 * 1024, eWhT[l], 256, 1024);
  tr(F(21), eOutT, 256, 64);

  // combined biases  (bi + bh)
  bs(F(5), F(6), nb[0]);
  for (int l = 0; l < 3; ++l) bs(F(9) + l * 1024, F(10) + l * 1024, nb[l + 1]);
  bs(F(15), F(16), eb[0]);
  for (int l = 0; l < 3; ++l) bs(F(19) + l * 1024, F(20) + l * 1024, eb[l + 1]);

  // latent context straight into the node concat buffer
  k_ctx<<<32, 256, 0, stream>>>(F(0), F(1), F(2), nodeX0);

  P.nWi0T = nWi0T; P.nWh0T = nWh0T; P.nOutT = nOutT;
  P.eWi0T = eWi0T; P.eWh0T = eWh0T; P.eOutT = eOutT;
  for (int l = 0; l < 3; ++l) {
    P.nWiT[l] = nWiT[l]; P.nWhT[l] = nWhT[l];
    P.eWiT[l] = eWiT[l]; P.eWhT[l] = eWhT[l];
  }
  for (int l = 0; l < 4; ++l) { P.nb[l] = nb[l]; P.eb[l] = eb[l]; }
  P.nOutb = F(12); P.eOutb = F(22);
  P.ncW0 = F(23); P.ncb0 = F(24);
  P.ncW1 = F(25); P.ncb1 = F(26);
  P.ncW2 = F(27); P.ncb2 = F(28);
  P.ecW0 = F(29); P.ecb0 = F(30);
  P.ecW1 = F(31); P.ecb1 = F(32);
  P.atomT = I(33); P.bondT = I(34);
  P.nodeX0 = nodeX0; P.edgeX0 = edgeX0;
  P.nodeH = nodeH; P.edgeH = edgeH; P.bufE = bufE;
  P.nodeC = nodeC; P.edgeC = edgeC;
  P.nodeEmb = nodeEmb; P.edgeEmb = edgeEmb;
  P.outLoss = (float*)d_out;

  grnn_main<<<1, 512, 0, stream>>>(P);
}
